// Synth_10522669875708
// MI455X (gfx1250) — matrix operand order repaired and checked by reading, not compiled, not run on hardware
//
#include <hip/hip_runtime.h>
#include <hip/hip_bf16.h>

// ---------------- problem constants ----------------
#define BB 2
#define FF 320
#define BLK 256
#define TT (FF*BLK)          // 81920 samples per batch
#define HARM 128
#define NCH 491
#define NPAD 512
#define KIN 80
#define KPAD 96
#define NCPAD 496
#define SRATE 24000.0f
#define TWO_PI 6.28318530718f

// channel offsets inside raw[.., 491]
#define CH_F0   0
#define CH_A    1
#define CH_AMP  2      // 128
#define CH_HMAG 130    // 256
#define CH_NMAG 386    // 80
#define CH_VR   466
#define CH_VD   467
#define CH_FFR  468    // 4
#define CH_FB   472    // 4
#define CH_FG   476    // 4
#define CH_OQ   480
#define CH_PC   481
#define CH_BR   482
#define CH_BSH  483    // 8

#define IRS_H 510      // irfft of 256 bins
#define IRS_N 158      // irfft of 80 bins
#define IRS_B 14       // irfft of 8 bins

// ---------------- workspace layout (floats) ----------------
#define WS_RAW   0u                       // 640*512
#define WS_F0C   (WS_RAW + 640u*512u)     // 640
#define WS_AMPS  (WS_F0C + 640u)          // 640*128
#define WS_VR    (WS_AMPS + 640u*128u)    // 640
#define WS_VD    (WS_VR + 640u)
#define WS_OQ    (WS_VD + 640u)
#define WS_PC    (WS_OQ + 640u)
#define WS_BR    (WS_PC + 640u)
#define WS_FFR   (WS_BR + 640u)           // 640*4
#define WS_FB    (WS_FFR + 2560u)         // 640*4
#define WS_FG    (WS_FB + 2560u)          // 640*4
#define WS_HIR   (WS_FG + 2560u)          // 640*510
#define WS_NIR   (WS_HIR + 640u*510u)     // 640*158
#define WS_BIR   (WS_NIR + 640u*158u)     // 640*14
#define WS_PHASE (WS_BIR + 640u*14u)      // 2*81920
#define WS_F0M   (WS_PHASE + 163840u)     // 2*81920
#define WS_HSIG  (WS_F0M + 163840u)       // 2*81920
#define WS_PACKA (WS_HSIG + 163840u)      // 640*96 halves  = 30720 floats
#define WS_PACKB (WS_PACKA + 30720u)      // 496*96 halves  = 23808 floats

typedef __attribute__((ext_vector_type(16))) _Float16 v16h;
typedef __attribute__((ext_vector_type(8)))  _Float16 v8h;
typedef __attribute__((ext_vector_type(8)))  float    v8f;

// ---------------- helpers ----------------
__device__ __forceinline__ float sigm(float x){ return 1.0f/(1.0f+__expf(-x)); }
__device__ __forceinline__ float scalefn(float x){
  float s = sigm(x);
  return 2.0f*__powf(s, 2.302585093f) + 1e-7f;   // 2*sigmoid(x)^ln(10)+1e-7
}
__device__ __forceinline__ unsigned hashu(unsigned x){
  x ^= x>>16; x *= 0x7feb352dU; x ^= x>>15; x *= 0x846ca68bU; x ^= x>>16; return x;
}
__device__ __forceinline__ float u01(unsigned x){            // [0,1)
  return (float)(hashu(x) >> 8) * (1.0f/16777216.0f);
}
__device__ __forceinline__ float lerpf(float a, float b, float t){ return a + (b-a)*t; }

// ---------------- 0) pack A (mel) and B (W^T), f32 -> zero-padded f16 ----------------
// Apack[row][k]  : 640 x 96, contiguous K per row
// Bpack[col][k]  : 496 x 96, contiguous K per column (W transposed)
__global__ __launch_bounds__(256)
void k_pack(const float* __restrict__ mel, const float* __restrict__ Wm,
            _Float16* __restrict__ apack, _Float16* __restrict__ bpack) {
  int idx = blockIdx.x * 256 + threadIdx.x;
  if (idx < 640 * KPAD) {
    int row = idx / KPAD, k = idx % KPAD;
    apack[idx] = (_Float16)((k < KIN) ? mel[row * KIN + k] : 0.0f);
  } else {
    idx -= 640 * KPAD;
    if (idx < NCPAD * KPAD) {
      int col = idx / KPAD, k = idx % KPAD;
      bpack[idx] = (_Float16)((k < KIN && col < NCH) ? Wm[k * NCH + col] : 0.0f);
    }
  }
}

// ---------------- 1) GEMM: raw = mel @ W + b via v_wmma_f32_16x16x32_f16 ----------------
// one wave32 per 16x16 C tile; fragments filled with aligned contiguous b128 loads
__global__ __launch_bounds__(32)
void k_gemm(const _Float16* __restrict__ apack, const _Float16* __restrict__ bpack,
            const float* __restrict__ bias, float* __restrict__ raw) {
  const int lane = threadIdx.x;
  const int n0 = blockIdx.x * 16;        // N tile over 496 (31 tiles)
  const int m0 = blockIdx.y * 16;        // M tile over 640 rows (40 tiles)
  const int hi = lane >> 4;              // half-wave selector
  const int l  = lane & 15;
  const _Float16* Ap = apack + (size_t)(m0 + l) * KPAD;  // row of A
  const _Float16* Bp = bpack + (size_t)(n0 + l) * KPAD;  // column of W
  v8f c = {};
  #pragma unroll
  for (int kt = 0; kt < KPAD; kt += 32) {
    if (kt + 32 < KPAD)                  // global_prefetch_b8 of next k-slab
      __builtin_prefetch(Bp + kt + 32, 0, 0);
    // A 16x32 f16 lane layout: e0..7 -> K=kt+hi*8+e ; e8..15 -> K=kt+16+hi*8+(e-8)
    v8h a_lo = *(const v8h*)(Ap + kt + hi * 8);
    v8h a_hi = *(const v8h*)(Ap + kt + 16 + hi * 8);
    v16h a = __builtin_shufflevector(a_lo, a_hi, 0,1,2,3,4,5,6,7,8,9,10,11,12,13,14,15);
    // B 32x16 f16 lane layout: lane&15 = N, e0..7 -> K=kt+hi*8+e ; e8..15 -> K=kt+16+hi*8+(e-8)
    v8h b_lo = *(const v8h*)(Bp + kt + hi * 8);
    v8h b_hi = *(const v8h*)(Bp + kt + 16 + hi * 8);
    v16h bfr = __builtin_shufflevector(b_lo, b_hi, 0,1,2,3,4,5,6,7,8,9,10,11,12,13,14,15);
    c = __builtin_amdgcn_wmma_f32_16x16x32_f16(false, a, false, bfr,
                                               (short)0, c, false, false);
  }
  const int col = n0 + l;
  if (col < NCH) {
    float bv = bias[col];
    #pragma unroll
    for (int r = 0; r < 8; ++r) {        // C layout: vgpr r -> M = hi*8+r, N = lane&15
      raw[(size_t)(m0 + hi * 8 + r) * NPAD + col] = c[r] + bv;
    }
  }
}

// ---------------- 2) per-frame activations ----------------
__global__ void k_act(const float* __restrict__ raw, float* __restrict__ ws,
                      float* __restrict__ f0_out) {
  int idx = blockIdx.x * blockDim.x + threadIdx.x;   // b*FF + f
  if (idx >= BB * FF) return;
  const float* r = raw + (size_t)idx * NPAD;
  // f0
  float s = sigm(r[CH_F0]);
  float f0 = 80.0f * __powf(12.5f, s);
  if (f0 < 80.0f) f0 = 0.0f;
  ws[WS_F0C + idx] = f0;
  f0_out[idx] = f0;
  // amplitudes (scale, normalize, * A)
  float A = scalefn(r[CH_A]);
  float sum = 0.0f;
  for (int h = 0; h < HARM; ++h) {
    float v = scalefn(r[CH_AMP + h]);
    ws[WS_AMPS + (size_t)idx * HARM + h] = v;
    sum += v;
  }
  float inv = A / (sum + 1e-8f);
  for (int h = 0; h < HARM; ++h) ws[WS_AMPS + (size_t)idx * HARM + h] *= inv;
  // scalar controls
  ws[WS_VR + idx] = sigm(r[CH_VR]) * 8.0f + 3.0f;
  ws[WS_VD + idx] = sigm(r[CH_VD]) * 0.5f;
  ws[WS_OQ + idx] = sigm(r[CH_OQ]) * 0.4f + 0.4f;
  ws[WS_PC + idx] = sigm(r[CH_PC]);
  ws[WS_BR + idx] = sigm(r[CH_BR]) * 0.5f;
  // formants
  const float base[4] = {500.0f, 1500.0f, 2500.0f, 3500.0f};
  float gmax = -1e30f, ge[4];
  for (int k = 0; k < 4; ++k) {
    ws[WS_FFR + idx*4 + k] = base[k] + tanhf(r[CH_FFR + k]) * 500.0f;
    ws[WS_FB  + idx*4 + k] = sigm(r[CH_FB + k]) * 200.0f + 50.0f;
    ge[k] = r[CH_FG + k];
    gmax = fmaxf(gmax, ge[k]);
  }
  float gs = 0.0f;
  for (int k = 0; k < 4; ++k) { ge[k] = __expf(ge[k] - gmax); gs += ge[k]; }
  for (int k = 0; k < 4; ++k) ws[WS_FG + idx*4 + k] = ge[k] / gs * 4.0f;
}

// running max over flattened (F*4) per batch
__global__ void k_cummax(float* __restrict__ ffr) {
  int b = threadIdx.x;
  if (b >= BB) return;
  float m = -1e30f;
  for (int f = 0; f < FF; ++f)
    for (int k = 0; k < 4; ++k) {
      int i = (b * FF + f) * 4 + k;
      m = fmaxf(m, ffr[i]);
      ffr[i] = m;
    }
}

// ---------------- 3) windowed IRs (direct inverse rDFT == irfft+roll+hann) ----------------
// mode 0: scale_fn(mags);  mode 1: softmax(-|x|) over nbins
__global__ __launch_bounds__(256)
void k_ir(const float* __restrict__ raw, float* __restrict__ ir,
          int nbins, int choff, int irs, int mode) {
  __shared__ float smag[256];
  int frame = blockIdx.x;                 // b*FF + f
  int tid = threadIdx.x;
  const float* r = raw + (size_t)frame * NPAD + choff;
  for (int i = tid; i < nbins; i += blockDim.x)
    smag[i] = (mode == 0) ? scalefn(r[i]) : __expf(-fabsf(r[i]));
  __syncthreads();
  if (mode == 1 && tid == 0) {
    float s2 = 0.0f;
    for (int i = 0; i < nbins; ++i) s2 += smag[i];
    float inv = 1.0f / s2;
    for (int i = 0; i < nbins; ++i) smag[i] *= inv;
  }
  __syncthreads();
  float irsf = (float)irs;
  for (int t = tid; t < irs; t += blockDim.x) {
    int u = t - irs / 2; if (u < 0) u += irs;       // roll by irs/2
    float acc = smag[0];
    for (int k = 1; k < nbins - 1; ++k)
      acc += 2.0f * smag[k] * __cosf(TWO_PI * (float)(k * u) / irsf);
    acc += smag[nbins - 1] * ((u & 1) ? -1.0f : 1.0f);   // Nyquist bin
    float win = 0.5f - 0.5f * __cosf(TWO_PI * (float)t / irsf);
    ir[(size_t)frame * irs + t] = (acc / irsf) * win;
  }
}

// ---------------- 4) vibrato + phase: two chained cumsums via workgroup scan ----------------
__global__ __launch_bounds__(1024)
void k_phase(const float* __restrict__ ws, const float* __restrict__ phi0p,
             float* __restrict__ phase, float* __restrict__ f0m,
             float* __restrict__ fp_out) {
  int b = blockIdx.x;
  int tid = threadIdx.x;
  __shared__ float sh[1024];
  __shared__ float carry1, carry2;
  if (tid == 0) { carry1 = 0.0f; carry2 = 0.0f; }
  __syncthreads();
  const float* f0c = ws + WS_F0C + b * FF;
  const float* vrc = ws + WS_VR  + b * FF;
  const float* vdc = ws + WS_VD  + b * FF;
  float phi0 = phi0p[b];
  for (int c0 = 0; c0 < TT; c0 += 1024) {
    int t = c0 + tid;
    int j = t >> 8, jp = min(j + 1, FF - 1);
    float fr = (float)(t & 255) * (1.0f / 256.0f);
    float vr = lerpf(vrc[j], vrc[jp], fr);
    float vd = lerpf(vdc[j], vdc[jp], fr);
    float pitch = lerpf(f0c[j], f0c[jp], fr);
    // ---- scan 1: cumsum(vrS) ----
    sh[tid] = vr; __syncthreads();
    for (int off = 1; off < 1024; off <<= 1) {
      float v = (tid >= off) ? sh[tid - off] : 0.0f;
      __syncthreads(); sh[tid] += v; __syncthreads();
    }
    float cum1 = sh[tid] + carry1;
    float tot1 = sh[1023] + carry1;
    __syncthreads();
    if (tid == 0) carry1 = tot1;
    // vibrato-modulated instantaneous f0
    float f0mv = pitch * (1.0f + vd * __sinf(TWO_PI * cum1 * (1.0f / SRATE)));
    // ---- scan 2: cumsum(f0m/SR) ----
    __syncthreads();
    sh[tid] = f0mv * (1.0f / SRATE); __syncthreads();
    for (int off = 1; off < 1024; off <<= 1) {
      float v = (tid >= off) ? sh[tid - off] : 0.0f;
      __syncthreads(); sh[tid] += v; __syncthreads();
    }
    float cum2 = sh[tid] + carry2;
    float tot2 = sh[1023] + carry2;
    __syncthreads();
    if (tid == 0) carry2 = tot2;
    __syncthreads();
    float ph = TWO_PI * cum2 + phi0;
    phase[b * TT + t] = ph;
    f0m[b * TT + t] = f0mv;
    if (t == TT - 1) fp_out[b] = fmodf(ph, TWO_PI);
  }
}

// ---------------- 5) harmonic bank: 128 sinusoids + formant envelope ----------------
__global__ __launch_bounds__(256)
void k_harm(const float* __restrict__ ws, float* __restrict__ hsig) {
  int blk = blockIdx.x;
  int b = blk / FF, f = blk % FF;
  int tid = threadIdx.x;
  int t = f * BLK + tid;
  int jp = min(f + 1, FF - 1);
  __shared__ float aJ[HARM], aJp[HARM], rph[HARM];
  if (tid < HARM) {
    aJ[tid]  = ws[WS_AMPS + ((size_t)(b * FF + f ) * HARM) + tid];
    aJp[tid] = ws[WS_AMPS + ((size_t)(b * FF + jp) * HARM) + tid];
    rph[tid] = u01(0x1337u + tid) * TWO_PI;   // fixed per-harmonic random phase
  }
  __syncthreads();
  float fr = (float)tid * (1.0f / 256.0f);
  int i0 = b * FF + f, i1 = b * FF + jp;
  float pcv = lerpf(ws[WS_PC + i0], ws[WS_PC + i1], fr);
  float oqv = lerpf(ws[WS_OQ + i0], ws[WS_OQ + i1], fr);
  float ffv[4], fbv[4], fgv[4];
  #pragma unroll
  for (int k = 0; k < 4; ++k) {
    ffv[k] = lerpf(ws[WS_FFR + i0*4 + k], ws[WS_FFR + i1*4 + k], fr);
    fbv[k] = lerpf(ws[WS_FB  + i0*4 + k], ws[WS_FB  + i1*4 + k], fr);
    fgv[k] = lerpf(ws[WS_FG  + i0*4 + k], ws[WS_FG  + i1*4 + k], fr);
  }
  float ph  = ws[WS_PHASE + (size_t)b * TT + t];
  float f0v = ws[WS_F0M   + (size_t)b * TT + t];
  float tiltc = -(1.0f - oqv) * 0.05f;
  float acc = 0.0f;
  for (int h = 0; h < HARM; ++h) {
    float n = (float)(h + 1);
    float freq = f0v * n;
    float env = 0.0f;
    #pragma unroll
    for (int k = 0; k < 4; ++k) {
      float z = (freq - ffv[k]) / fbv[k];
      env += fgv[k] * __expf(-0.5f * z * z);
    }
    float tilt = __expf(tiltc * (float)h);
    float amp = lerpf(aJ[h], aJp[h], fr);
    float mask = (freq < SRATE * 0.5f) ? 1.0f : 0.0f;
    acc += amp * (1.0f + env) * tilt * mask * __sinf(ph * n + (1.0f - pcv) * rph[h]);
  }
  hsig[(size_t)b * TT + t] = acc;
}

// ---------------- 6) breath noise: short filtered noise added into hsig ----------------
__global__ __launch_bounds__(256)
void k_breath(const float* __restrict__ ws, float* __restrict__ hsig) {
  int idx = blockIdx.x * blockDim.x + threadIdx.x;
  int b = idx / TT, t = idx % TT;
  int j = t >> 8, jp = min(j + 1, FF - 1);
  float fr = (float)(t & 255) * (1.0f / 256.0f);
  float brv = lerpf(ws[WS_BR + b * FF + j], ws[WS_BR + b * FF + jp], fr);
  float acc = 0.0f;
  #pragma unroll
  for (int tau = 0; tau < IRS_B; ++tau) {
    int s = t + IRS_B / 2 - tau;
    if (s >= 0 && s < TT) {
      float x = u01((unsigned)(b * TT + s) * 2654435769u + 7u) * 2.0f - 1.0f;
      acc += x * ws[WS_BIR + ((size_t)(b * FF + (s >> 8))) * IRS_B + tau];
    }
  }
  hsig[(size_t)b * TT + t] += acc * brv;
}

// ---------------- 7) final: harmonic filter + noise filter + sum ----------------
__global__ __launch_bounds__(256)
void k_final(const float* __restrict__ ws,
             float* __restrict__ o_sig, float* __restrict__ o_harm,
             float* __restrict__ o_noise) {
  int idx = blockIdx.x * blockDim.x + threadIdx.x;
  int b = idx / TT, t = idx % TT;
  const float* hsig = ws + WS_HSIG + (size_t)b * TT;
  // time-varying conv with per-frame 510-tap IR (== FFT overlap-add in reference)
  float hacc = 0.0f;
  for (int tau = 0; tau < IRS_H; ++tau) {
    int s = t + IRS_H / 2 - tau;
    if (s >= 0 && s < TT)
      hacc += hsig[s] * ws[WS_HIR + ((size_t)(b * FF + (s >> 8))) * IRS_H + tau];
  }
  float nacc = 0.0f;
  for (int tau = 0; tau < IRS_N; ++tau) {
    int s = t + IRS_N / 2 - tau;
    if (s >= 0 && s < TT) {
      float x = u01((unsigned)(b * TT + s) * 2246822519u + 42u) * 2.0f - 1.0f;
      nacc += x * ws[WS_NIR + ((size_t)(b * FF + (s >> 8))) * IRS_N + tau];
    }
  }
  o_harm[idx]  = hacc;
  o_noise[idx] = nacc;
  o_sig[idx]   = hacc + nacc;
}

// ---------------- launch ----------------
extern "C" void kernel_launch(void* const* d_in, const int* in_sizes, int n_in,
                              void* d_out, int out_size, void* d_ws, size_t ws_size,
                              hipStream_t stream) {
  const float* mel  = (const float*)d_in[0];   // (2,320,80)
  const float* phi0 = (const float*)d_in[1];   // (2,1,1)
  const float* Wm   = (const float*)d_in[2];   // (80,491)
  const float* bias = (const float*)d_in[3];   // (491,)
  float* out = (float*)d_out;
  float* o_sig   = out;                         // (2,81920)
  float* o_f0    = out + 163840;                // (2,320,1)
  float* o_fp    = out + 164480;                // (2,1,1)
  float* o_harm  = out + 164482;                // (2,81920)
  float* o_noise = out + 328322;                // (2,81920)
  float* ws = (float*)d_ws;
  float* raw = ws + WS_RAW;
  _Float16* apack = (_Float16*)(ws + WS_PACKA);
  _Float16* bpack = (_Float16*)(ws + WS_PACKB);

  // 0) pack operands to padded f16 (A row-major, B transposed/column-major)
  k_pack<<<(640 * KPAD + NCPAD * KPAD + 255) / 256, 256, 0, stream>>>(mel, Wm, apack, bpack);
  // 1) WMMA GEMM: raw = mel @ W + b
  k_gemm<<<dim3(NCPAD / 16, (BB * FF) / 16), 32, 0, stream>>>(apack, bpack, bias, raw);
  // 2) activations
  k_act<<<(BB * FF + 255) / 256, 256, 0, stream>>>(raw, ws, o_f0);
  k_cummax<<<1, 32, 0, stream>>>(ws + WS_FFR);
  // 3) windowed IRs for the three filters
  k_ir<<<BB * FF, 256, 0, stream>>>(raw, ws + WS_HIR, 256, CH_HMAG, IRS_H, 0);
  k_ir<<<BB * FF, 256, 0, stream>>>(raw, ws + WS_NIR,  80, CH_NMAG, IRS_N, 0);
  k_ir<<<BB * FF, 256, 0, stream>>>(raw, ws + WS_BIR,   8, CH_BSH,  IRS_B, 1);
  // 4) vibrato + phase accumulation (chained scans)
  k_phase<<<BB, 1024, 0, stream>>>(ws, phi0, ws + WS_PHASE, ws + WS_F0M, o_fp);
  // 5) harmonic bank
  k_harm<<<BB * FF, 256, 0, stream>>>(ws, ws + WS_HSIG);
  // 6) breath
  k_breath<<<(BB * TT) / 256, 256, 0, stream>>>(ws, ws + WS_HSIG);
  // 7) final filters + mix
  k_final<<<(BB * TT) / 256, 256, 0, stream>>>(ws, o_sig, o_harm, o_noise);
}
